// HALLinear_84902913507492
// MI455X (gfx1250) — matrix operand order repaired and checked by reading, not compiled, not run on hardware
//
#include <hip/hip_runtime.h>

#define M_DIM 8192
#define N_DIM 16384
#define K_DIM 4096

#define BM 128
#define BN 256
#define BK 32
#define SA 40                 // padded LDS K-stride (elements): 80B keeps 16B align, conflict-free
#define KT (K_DIM / BK)       // 128 K-steps

typedef __attribute__((ext_vector_type(16))) __bf16 v16bf;
typedef __attribute__((ext_vector_type(8)))  __bf16 v8bf;
typedef __attribute__((ext_vector_type(4)))  __bf16 v4bf;
typedef __attribute__((ext_vector_type(8)))  float  v8f;
typedef __attribute__((ext_vector_type(4)))  float  f4;   // 16B-aligned -> global_load_b128

union V16 { v16bf v; v8bf h[2]; };

// LDS-only barrier: drain DS, signal+wait workgroup barrier. Unlike __syncthreads()
// this does NOT wait out in-flight global loads, so the k+1 prefetch stays live
// across the barrier and is drained only at its cvt consumers.
__device__ __forceinline__ void lds_barrier() {
    asm volatile("s_wait_dscnt 0x0" ::: "memory");
    asm volatile("s_barrier_signal -1\n\t"
                 "s_barrier_wait -1" ::: "memory");
}

__global__ __launch_bounds__(256) void hal_linear_wmma_bf16(
    const float* __restrict__ X,   // [M, K] fp32
    const float* __restrict__ W,   // [N, K] fp32
    const float* __restrict__ Bv,  // [N]    fp32
    float* __restrict__ Y)         // [M, N] fp32
{
    __shared__ __bf16 As[2][BM * SA];   // 128 x 32 bf16 (padded), x2 buffers
    __shared__ __bf16 Bs[2][BN * SA];   // 256 x 32 bf16 (padded), x2 buffers

    const int t    = threadIdx.x;
    const int wave = t >> 5;
    const int lane = t & 31;
    const int lm   = lane & 15;
    const int g    = lane >> 4;        // half-wave group (0/1)
    const int wm   = wave & 1;         // 2 waves along M -> 64-row strips
    const int wn   = wave >> 1;        // 4 waves along N -> 64-col strips

    const int m0 = blockIdx.y * BM;
    const int n0 = blockIdx.x * BN;

    const char* Xc = (const char*)X;
    const char* Wc = (const char*)W;

    // ---- Precompute global byte offsets (u32) and LDS element offsets ----
    // A tile: 128 rows x 8 f4-cols = 1024 slots, 4 per thread.
    unsigned baX[4]; int ldsA[4];
#pragma unroll
    for (int j = 0; j < 4; ++j) {
        const int li  = t + j * 256;
        const int row = li >> 3;
        const int c4  = li & 7;
        baX[j]  = (unsigned)(((m0 + row) * K_DIM + c4 * 4) * 4);
        ldsA[j] = row * SA + c4 * 4;
    }
    // B tile: 256 rows x 8 f4-cols = 2048 slots, 8 per thread.
    unsigned baW[8]; int ldsB[8];
#pragma unroll
    for (int j = 0; j < 8; ++j) {
        const int li  = t + j * 256;
        const int row = li >> 3;
        const int c4  = li & 7;
        baW[j]  = (unsigned)(((n0 + row) * K_DIM + c4 * 4) * 4);
        ldsB[j] = row * SA + c4 * 4;
    }

    // Fragment LDS element offsets (per CDNA5 ISA 7.12.2 16-bit A/B layouts).
    int aof0[4], aof1[4], bof0[4], bof1[4];
#pragma unroll
    for (int mt = 0; mt < 4; ++mt) {
        const int m = wm * 64 + mt * 16 + lm;
        aof0[mt] = m * SA + 8 * g;         // halves 0-7 : K =  8g .. 8g+7
        aof1[mt] = m * SA + 16 + 8 * g;    // halves 8-15: K = 16+8g ..
    }
#pragma unroll
    for (int nt = 0; nt < 4; ++nt) {
        const int n = wn * 64 + nt * 16 + lm;
        bof0[nt] = n * SA + 8 * g;         // halves 0-7 : K =  8g .. 8g+7
        bof1[nt] = n * SA + 16 + 8 * g;    // halves 8-15: K = 16+8g ..
    }

    f4 ra[4], rb[8];

    auto load_tiles = [&](int kt) {
        const unsigned koff = (unsigned)(kt * BK * 4);   // 128B per K-step
#pragma unroll
        for (int j = 0; j < 4; ++j) ra[j] = *(const f4*)(Xc + (baX[j] + koff));
#pragma unroll
        for (int j = 0; j < 8; ++j) rb[j] = *(const f4*)(Wc + (baW[j] + koff));
    };

    auto store_tiles = [&](int buf) {
#pragma unroll
        for (int j = 0; j < 4; ++j) {
            v4bf p = { (__bf16)ra[j].x, (__bf16)ra[j].y, (__bf16)ra[j].z, (__bf16)ra[j].w };
            *(v4bf*)&As[buf][ldsA[j]] = p;
        }
#pragma unroll
        for (int j = 0; j < 8; ++j) {
            v4bf p = { (__bf16)rb[j].x, (__bf16)rb[j].y, (__bf16)rb[j].z, (__bf16)rb[j].w };
            *(v4bf*)&Bs[buf][ldsB[j]] = p;
        }
    };

    v8f acc[4][4];
#pragma unroll
    for (int mt = 0; mt < 4; ++mt)
#pragma unroll
        for (int nt = 0; nt < 4; ++nt)
#pragma unroll
            for (int i = 0; i < 8; ++i)
                acc[mt][nt][i] = 0.0f;

    load_tiles(0);
    store_tiles(0);
    lds_barrier();

    for (int kt = 0; kt < KT; ++kt) {
        const int cur = kt & 1;

        // Prefetch next K-tile (stays in flight across barriers now).
        if (kt + 1 < KT) load_tiles(kt + 1);

        V16 afr[4], bfr[4];
#pragma unroll
        for (int mt = 0; mt < 4; ++mt) {
            afr[mt].h[0] = *(v8bf*)&As[cur][aof0[mt]];
            afr[mt].h[1] = *(v8bf*)&As[cur][aof1[mt]];
        }
#pragma unroll
        for (int nt = 0; nt < 4; ++nt) {
            bfr[nt].h[0] = *(v8bf*)&Bs[cur][bof0[nt]];
            bfr[nt].h[1] = *(v8bf*)&Bs[cur][bof1[nt]];
        }

#pragma unroll
        for (int mt = 0; mt < 4; ++mt)
#pragma unroll
            for (int nt = 0; nt < 4; ++nt)
                acc[mt][nt] = __builtin_amdgcn_wmma_f32_16x16x32_bf16(
                    false, afr[mt].v, false, bfr[nt].v,
                    (short)0, acc[mt][nt], false, false);

        lds_barrier();                     // all waves done reading buf[cur]
        if (kt + 1 < KT) {
            store_tiles(cur ^ 1);          // cvt fp32->bf16, ds_store other buffer
            lds_barrier();
        }
    }

    // Epilogue: C/D layout VGPR r -> M = r + 8*(lane/16), N = lane%16. Add bias.
#pragma unroll
    for (int nt = 0; nt < 4; ++nt) {
        const int col = n0 + wn * 64 + nt * 16 + lm;
        const float bval = Bv[col];
#pragma unroll
        for (int mt = 0; mt < 4; ++mt) {
            const int mbase = m0 + wm * 64 + mt * 16 + 8 * g;
#pragma unroll
            for (int r = 0; r < 8; ++r) {
                Y[(mbase + r) * N_DIM + col] = acc[mt][nt][r] + bval;
            }
        }
    }
}

extern "C" void kernel_launch(void* const* d_in, const int* in_sizes, int n_in,
                              void* d_out, int out_size, void* d_ws, size_t ws_size,
                              hipStream_t stream) {
    const float* x = (const float*)d_in[0];   // [8192, 4096]
    const float* w = (const float*)d_in[1];   // [16384, 4096]
    const float* b = (const float*)d_in[2];   // [16384]
    float* y = (float*)d_out;                 // [8192, 16384]

    dim3 grid(N_DIM / BN, M_DIM / BM);        // (64, 64)
    dim3 block(256);
    hipLaunchKernelGGL(hal_linear_wmma_bf16, grid, block, 0, stream, x, w, b, y);
}
